// GATEncoder_7241314861489
// MI455X (gfx1250) — matrix operand order repaired and checked by reading, not compiled, not run on hardware
//
#include <hip/hip_runtime.h>
#include <hip/hip_bf16.h>
#include <cstdint>
#include <cstddef>

// ---------------------------------------------------------------------------
// Types for CDNA5 WMMA (gfx1250, wave32)
// ---------------------------------------------------------------------------
typedef __attribute__((ext_vector_type(16))) __bf16 v16bf;
typedef __attribute__((ext_vector_type(8)))  __bf16 v8bf;
typedef __attribute__((ext_vector_type(8)))  float  v8f;

#define NEG_SLOPE 0.2f
#define SM_EPS    1e-16f

// order-preserving float<->uint key for atomic max on floats
// key(-inf) = ~0xFF800000 = 0x007FFFFF
#define NEGINF_KEY 0x007FFFFFu

__device__ __forceinline__ unsigned fkey(float f) {
  unsigned u = __float_as_uint(f);
  return (u & 0x80000000u) ? ~u : (u | 0x80000000u);
}
__device__ __forceinline__ float keyf(unsigned k) {
  unsigned u = (k & 0x80000000u) ? (k & 0x7fffffffu) : ~k;
  return __uint_as_float(u);
}
__device__ __forceinline__ void atomAddF(float* p, float v) {
  (void)__hip_atomic_fetch_add(p, v, __ATOMIC_RELAXED, __HIP_MEMORY_SCOPE_AGENT);
}

// ---------------------------------------------------------------------------
// Elementwise helpers
// ---------------------------------------------------------------------------
__global__ void k_f2bf(const float* __restrict__ in, __bf16* __restrict__ out, int n) {
  int t = blockIdx.x * blockDim.x + threadIdx.x;
  if (t < n) out[t] = (__bf16)in[t];
}

__global__ void k_fill_f32(float* __restrict__ p, float v, int n) {
  int t = blockIdx.x * blockDim.x + threadIdx.x;
  if (t < n) p[t] = v;
}

__global__ void k_fill_u32(unsigned* __restrict__ p, unsigned v, int n) {
  int t = blockIdx.x * blockDim.x + threadIdx.x;
  if (t < n) p[t] = v;
}

// Pack a row-major fp32 weight [K,N] into bf16 WMMA B fragments.
// Fragment block b = kt*(N/16)+nt is 512 halves: lane L (0..31) x elem e (0..15)
//   column n = nt*16 + (L%16), K = kt*32 + (L<16 ? 0 : 8) + (e<8 ? e : 8+e)
__global__ void k_pack_b(const float* __restrict__ W, __bf16* __restrict__ Bp, int K, int N) {
  int t = blockIdx.x * blockDim.x + threadIdx.x;
  if (t >= K * N) return;
  int Ntiles = N >> 4;
  int b = t >> 9;        // / 512
  int r = t & 511;
  int L = r >> 4;
  int e = r & 15;
  int kt = b / Ntiles, nt = b % Ntiles;
  int k = kt * 32 + ((L < 16) ? 0 : 8) + ((e < 8) ? e : 8 + e);
  int n = nt * 16 + (L & 15);
  Bp[t] = (__bf16)W[(size_t)k * N + n];
}

// A-fragment loader: lane L row m=L%16; elems 0..7 -> K = kb+e,
// elems 8..15 -> K = 16+kb+(e-8), kb = (L<16)?0:8.
__device__ __forceinline__ v16bf load_a_frag(const __bf16* __restrict__ Arow,
                                             int ki, int kb) {
  v8bf a0 = *(const v8bf*)(Arow + ki + kb);
  v8bf a1 = *(const v8bf*)(Arow + ki + 16 + kb);
  return __builtin_shufflevector(a0, a1, 0, 1, 2, 3, 4, 5, 6, 7,
                                 8, 9, 10, 11, 12, 13, 14, 15);
}

// ---------------------------------------------------------------------------
// WMMA GEMM: C[M,N] (f32) = A[M,K] (bf16 row-major) * Bp (packed bf16 frags)
// One wave per 16x64 output strip: grid = (M/16, N/64), block = 32.
// Software-pipelined: iteration i's WMMAs consume fragments loaded in
// iteration i-1; loads for i+1 are in flight during the 4 WMMAs, so the
// s_wait before the WMMA group covers only already-landed loads.
// ---------------------------------------------------------------------------
__global__ __launch_bounds__(32) void k_wmma_gemm_bf16(
    const __bf16* __restrict__ A, const __bf16* __restrict__ Bp,
    float* __restrict__ C, int M, int N, int K) {
  const int lane = threadIdx.x;
  const int row0 = blockIdx.x << 4;
  const int ntBase = blockIdx.y << 2;   // 4 col tiles per wave
  const int Ntiles = N >> 4;
  const int m = row0 + (lane & 15);
  const int kb = (lane < 16) ? 0 : 8;
  const __bf16* Arow = A + (size_t)m * K;
  const size_t bstride = (size_t)Ntiles * 512;   // halves per K-step
  const __bf16* bptr = Bp + (size_t)ntBase * 512 + lane * 16;

  v8f acc0 = (v8f){0.f, 0.f, 0.f, 0.f, 0.f, 0.f, 0.f, 0.f};
  v8f acc1 = acc0, acc2 = acc0, acc3 = acc0;

  // pipeline prologue: fragments for ki = 0
  v16bf aC = load_a_frag(Arow, 0, kb);
  v16bf b0 = *(const v16bf*)(bptr);
  v16bf b1 = *(const v16bf*)(bptr + 512);
  v16bf b2 = *(const v16bf*)(bptr + 1024);
  v16bf b3 = *(const v16bf*)(bptr + 1536);

  for (int ki = 32; ki < K; ki += 32) {
    bptr += bstride;
    if (ki + 32 < K) __builtin_prefetch(Arow + ki + 32, 0, 3);
    // issue next-step loads BEFORE consuming current fragments
    v16bf aN = load_a_frag(Arow, ki, kb);
    v16bf n0 = *(const v16bf*)(bptr);
    v16bf n1 = *(const v16bf*)(bptr + 512);
    v16bf n2 = *(const v16bf*)(bptr + 1024);
    v16bf n3 = *(const v16bf*)(bptr + 1536);

    acc0 = __builtin_amdgcn_wmma_f32_16x16x32_bf16(false, aC, false, b0, (short)0, acc0, false, false);
    acc1 = __builtin_amdgcn_wmma_f32_16x16x32_bf16(false, aC, false, b1, (short)0, acc1, false, false);
    acc2 = __builtin_amdgcn_wmma_f32_16x16x32_bf16(false, aC, false, b2, (short)0, acc2, false, false);
    acc3 = __builtin_amdgcn_wmma_f32_16x16x32_bf16(false, aC, false, b3, (short)0, acc3, false, false);

    aC = aN; b0 = n0; b1 = n1; b2 = n2; b3 = n3;
  }

  // epilogue: last K-step
  acc0 = __builtin_amdgcn_wmma_f32_16x16x32_bf16(false, aC, false, b0, (short)0, acc0, false, false);
  acc1 = __builtin_amdgcn_wmma_f32_16x16x32_bf16(false, aC, false, b1, (short)0, acc1, false, false);
  acc2 = __builtin_amdgcn_wmma_f32_16x16x32_bf16(false, aC, false, b2, (short)0, acc2, false, false);
  acc3 = __builtin_amdgcn_wmma_f32_16x16x32_bf16(false, aC, false, b3, (short)0, acc3, false, false);

  // C/D layout: VGPR r -> M = r (+8 for lanes 16..31), col = lane%16
  const int mrow = row0 + ((lane < 16) ? 0 : 8);
  float* Cp = C + (size_t)mrow * N + (ntBase << 4) + (lane & 15);
#pragma unroll
  for (int r = 0; r < 8; ++r) {
    Cp[(size_t)r * N + 0]  = acc0[r];
    Cp[(size_t)r * N + 16] = acc1[r];
    Cp[(size_t)r * N + 32] = acc2[r];
    Cp[(size_t)r * N + 48] = acc3[r];
  }
}

// ---------------------------------------------------------------------------
// Per-node attention scores: sd[i,h] = dot(xp[i,h,:], att[h,:C]),
//                            ss[i,h] = dot(xp[i,h,:], att[h,C:2C])
// One wave32 per (node, head).
// ---------------------------------------------------------------------------
__global__ void k_scores(const float* __restrict__ xp, const float* __restrict__ att,
                         float* __restrict__ sd, float* __restrict__ ss,
                         int N, int H, int C) {
  int wave = blockIdx.x * (blockDim.x >> 5) + (threadIdx.x >> 5);
  int lane = threadIdx.x & 31;
  if (wave >= N * H) return;
  int node = wave / H, head = wave % H;
  const float* xrow = xp + (size_t)node * H * C + (size_t)head * C;
  const float* ad = att + (size_t)head * 2 * C;
  const float* as = ad + C;
  float d = 0.f, s = 0.f;
  for (int c = lane; c < C; c += 32) {
    float v = xrow[c];
    d += v * ad[c];
    s += v * as[c];
  }
#pragma unroll
  for (int off = 16; off > 0; off >>= 1) {
    d += __shfl_xor(d, off, 32);
    s += __shfl_xor(s, off, 32);
  }
  if (lane == 0) {
    sd[wave] = d;
    ss[wave] = s;
  }
}

// alpha[e,h] = leaky_relu(sd[dst,h] + ss[src,h]); keyed atomicMax into amaxU
__global__ void k_edge_alpha(const int* __restrict__ src, const int* __restrict__ dst,
                             const float* __restrict__ sd, const float* __restrict__ ss,
                             float* __restrict__ alpha, unsigned* __restrict__ amaxU,
                             int EH, int H) {
  int t = blockIdx.x * blockDim.x + threadIdx.x;
  if (t >= EH) return;
  int e = t / H, h = t % H;
  float a = sd[dst[e] * H + h] + ss[src[e] * H + h];
  a = (a >= 0.f) ? a : NEG_SLOPE * a;
  alpha[t] = a;
  atomicMax(&amaxU[dst[e] * H + h], fkey(a));
}

__global__ void k_amax_fin(const unsigned* __restrict__ u, float* __restrict__ f, int n) {
  int t = blockIdx.x * blockDim.x + threadIdx.x;
  if (t >= n) return;
  float v = keyf(u[t]);
  f[t] = __builtin_isfinite(v) ? v : 0.f;
}

// ex[e,h] = exp(alpha - amax[dst,h]); atomicAdd into denom[dst,h]
__global__ void k_edge_exp(const int* __restrict__ dst, float* __restrict__ alphaEx,
                           const float* __restrict__ amaxF, float* __restrict__ denom,
                           int EH, int H) {
  int t = blockIdx.x * blockDim.x + threadIdx.x;
  if (t >= EH) return;
  int e = t / H, h = t % H;
  float ex = expf(alphaEx[t] - amaxF[dst[e] * H + h]);
  alphaEx[t] = ex;
  atomAddF(&denom[dst[e] * H + h], ex);
}

// Wave per edge: out[dst, h, c] += xp[src, h, c] * a(e,h) * ew[e]
__global__ void k_aggregate(const int* __restrict__ src, const int* __restrict__ dst,
                            const float* __restrict__ ew, const float* __restrict__ xp,
                            const float* __restrict__ ex, const float* __restrict__ denom,
                            float* __restrict__ out, int E, int H, int C, int cshift) {
  int e = blockIdx.x * (blockDim.x >> 5) + (threadIdx.x >> 5);
  int lane = threadIdx.x & 31;
  if (e >= E) return;
  int s = src[e], d = dst[e];
  float w = ew[e];
  float ah[4];
#pragma unroll
  for (int h = 0; h < 4; ++h)
    ah[h] = (h < H) ? (ex[e * H + h] / (denom[d * H + h] + SM_EPS)) * w : 0.f;
  int HC = H * C;
  const float* xs = xp + (size_t)s * HC;
  float* od = out + (size_t)d * HC;
  for (int c = lane; c < HC; c += 32)
    atomAddF(&od[c], xs[c] * ah[c >> cshift]);
}

// h = elu(agg + bias), write bf16 copy for the next GEMM
__global__ void k_bias_elu_bf16(const float* __restrict__ agg, const float* __restrict__ bias,
                                __bf16* __restrict__ hbf, int n, int HC) {
  int t = blockIdx.x * blockDim.x + threadIdx.x;
  if (t >= n) return;
  float v = agg[t] + bias[t % HC];
  v = (v > 0.f) ? v : (expf(v) - 1.f);
  hbf[t] = (__bf16)v;
}

__global__ void k_bias_out(float* __restrict__ out, const float* __restrict__ agg,
                           const float* __restrict__ bias, int n, int C) {
  int t = blockIdx.x * blockDim.x + threadIdx.x;
  if (t >= n) return;
  out[t] = agg[t] + bias[t % C];
}

// ---------------------------------------------------------------------------
// Host launcher
// ---------------------------------------------------------------------------
static inline char* carve(char*& p, size_t bytes) {
  char* r = p;
  p += (bytes + 255) & ~(size_t)255;
  return r;
}
static inline int cdiv(int a, int b) { return (a + b - 1) / b; }

extern "C" void kernel_launch(void* const* d_in, const int* in_sizes, int n_in,
                              void* d_out, int out_size, void* d_ws, size_t ws_size,
                              hipStream_t stream) {
  (void)n_in; (void)out_size; (void)ws_size;
  const int IN = 512, HID = 256, LAT = 128, H1 = 4, C1 = 64;
  const int N = in_sizes[0] / IN;
  const int E = in_sizes[2];

  const float* x    = (const float*)d_in[0];
  const int*   ei   = (const int*)d_in[1];
  const int*   src  = ei;
  const int*   dst  = ei + E;
  const float* ew   = (const float*)d_in[2];
  const float* W1   = (const float*)d_in[3];
  const float* att1 = (const float*)d_in[4];
  const float* b1   = (const float*)d_in[5];
  const float* Wmu  = (const float*)d_in[6];
  const float* attm = (const float*)d_in[7];
  const float* bmu  = (const float*)d_in[8];
  const float* Wlv  = (const float*)d_in[9];
  const float* attl = (const float*)d_in[10];
  const float* blv  = (const float*)d_in[11];
  float* out = (float*)d_out;

  // ---- workspace layout --------------------------------------------------
  char* p = (char*)d_ws;
  __bf16* xbf   = (__bf16*)carve(p, (size_t)N * IN * 2);
  __bf16* W1p   = (__bf16*)carve(p, (size_t)IN * HID * 2);
  __bf16* Wmup  = (__bf16*)carve(p, (size_t)HID * LAT * 2);
  __bf16* Wlvp  = (__bf16*)carve(p, (size_t)HID * LAT * 2);
  float*  xp1   = (float*)carve(p, (size_t)N * HID * 4);
  float*  sd1   = (float*)carve(p, (size_t)N * H1 * 4);
  float*  ss1   = (float*)carve(p, (size_t)N * H1 * 4);
  unsigned* amU = (unsigned*)carve(p, (size_t)N * H1 * 4);
  float*  amF   = (float*)carve(p, (size_t)N * H1 * 4);
  float*  den   = (float*)carve(p, (size_t)N * H1 * 4);
  float*  aex   = (float*)carve(p, (size_t)E * H1 * 4);
  float*  agg1  = (float*)carve(p, (size_t)N * HID * 4);
  __bf16* hbf   = (__bf16*)carve(p, (size_t)N * HID * 2);
  float*  xpmu  = (float*)carve(p, (size_t)N * LAT * 4);
  float*  xplv  = (float*)carve(p, (size_t)N * LAT * 4);
  float*  agg2  = (float*)carve(p, (size_t)N * LAT * 4);

  const int TB = 256;

  // ---- stage 0: conversions / weight packing -----------------------------
  k_f2bf<<<cdiv(N * IN, TB), TB, 0, stream>>>(x, xbf, N * IN);
  k_pack_b<<<cdiv(IN * HID, TB), TB, 0, stream>>>(W1, W1p, IN, HID);
  k_pack_b<<<cdiv(HID * LAT, TB), TB, 0, stream>>>(Wmu, Wmup, HID, LAT);
  k_pack_b<<<cdiv(HID * LAT, TB), TB, 0, stream>>>(Wlv, Wlvp, HID, LAT);

  // ---- layer 1: xp1 = x @ W1 (WMMA bf16 -> f32) --------------------------
  k_wmma_gemm_bf16<<<dim3(N / 16, HID / 64), 32, 0, stream>>>(xbf, W1p, xp1, N, HID, IN);

  // attention layer 1 (H=4, C=64)
  k_scores<<<cdiv(N * H1 * 32, TB), TB, 0, stream>>>(xp1, att1, sd1, ss1, N, H1, C1);
  k_fill_u32<<<cdiv(N * H1, TB), TB, 0, stream>>>(amU, NEGINF_KEY, N * H1);
  k_fill_f32<<<cdiv(N * H1, TB), TB, 0, stream>>>(den, 0.f, N * H1);
  k_edge_alpha<<<cdiv(E * H1, TB), TB, 0, stream>>>(src, dst, sd1, ss1, aex, amU, E * H1, H1);
  k_amax_fin<<<cdiv(N * H1, TB), TB, 0, stream>>>(amU, amF, N * H1);
  k_edge_exp<<<cdiv(E * H1, TB), TB, 0, stream>>>(dst, aex, amF, den, E * H1, H1);
  k_fill_f32<<<cdiv(N * HID, TB), TB, 0, stream>>>(agg1, 0.f, N * HID);
  k_aggregate<<<cdiv(E * 32, TB), TB, 0, stream>>>(src, dst, ew, xp1, aex, den, agg1, E, H1, C1, 6);
  k_bias_elu_bf16<<<cdiv(N * HID, TB), TB, 0, stream>>>(agg1, b1, hbf, N * HID, HID);

  // ---- layer 2: xpmu / xplv = h @ Wmu / Wlv ------------------------------
  k_wmma_gemm_bf16<<<dim3(N / 16, LAT / 64), 32, 0, stream>>>(hbf, Wmup, xpmu, N, LAT, HID);
  k_wmma_gemm_bf16<<<dim3(N / 16, LAT / 64), 32, 0, stream>>>(hbf, Wlvp, xplv, N, LAT, HID);

  // mu head (H=1, C=128)
  k_scores<<<cdiv(N * 32, TB), TB, 0, stream>>>(xpmu, attm, sd1, ss1, N, 1, LAT);
  k_fill_u32<<<cdiv(N, TB), TB, 0, stream>>>(amU, NEGINF_KEY, N);
  k_fill_f32<<<cdiv(N, TB), TB, 0, stream>>>(den, 0.f, N);
  k_edge_alpha<<<cdiv(E, TB), TB, 0, stream>>>(src, dst, sd1, ss1, aex, amU, E, 1);
  k_amax_fin<<<cdiv(N, TB), TB, 0, stream>>>(amU, amF, N);
  k_edge_exp<<<cdiv(E, TB), TB, 0, stream>>>(dst, aex, amF, den, E, 1);
  k_fill_f32<<<cdiv(N * LAT, TB), TB, 0, stream>>>(agg2, 0.f, N * LAT);
  k_aggregate<<<cdiv(E * 32, TB), TB, 0, stream>>>(src, dst, ew, xpmu, aex, den, agg2, E, 1, LAT, 7);
  k_bias_out<<<cdiv(N * LAT, TB), TB, 0, stream>>>(out, agg2, bmu, N * LAT, LAT);

  // logvar head
  k_scores<<<cdiv(N * 32, TB), TB, 0, stream>>>(xplv, attl, sd1, ss1, N, 1, LAT);
  k_fill_u32<<<cdiv(N, TB), TB, 0, stream>>>(amU, NEGINF_KEY, N);
  k_fill_f32<<<cdiv(N, TB), TB, 0, stream>>>(den, 0.f, N);
  k_edge_alpha<<<cdiv(E, TB), TB, 0, stream>>>(src, dst, sd1, ss1, aex, amU, E, 1);
  k_amax_fin<<<cdiv(N, TB), TB, 0, stream>>>(amU, amF, N);
  k_edge_exp<<<cdiv(E, TB), TB, 0, stream>>>(dst, aex, amF, den, E, 1);
  k_fill_f32<<<cdiv(N * LAT, TB), TB, 0, stream>>>(agg2, 0.f, N * LAT);
  k_aggregate<<<cdiv(E * 32, TB), TB, 0, stream>>>(src, dst, ew, xplv, aex, den, agg2, E, 1, LAT, 7);
  k_bias_out<<<cdiv(N * LAT, TB), TB, 0, stream>>>(out + (size_t)N * LAT, agg2, blv, N * LAT, LAT);
}
